// ReplicatorDerivLayer_17523466568260
// MI455X (gfx1250) — matrix operand order repaired and checked by reading, not compiled, not run on hardware
//
#include <hip/hip_runtime.h>

// ---------------------------------------------------------------------------
// ReplicatorDerivLayer for MI455X (gfx1250, wave32, WMMA).
// Pipeline (B=8, E=512, S=2048):
//   K0: cast x, W1, W2 to bf16 (b128 vector path)
//   K1: F1 = W1^T X      (bf16 WMMA, f32 accum)
//   K2: F2 = W2   X
//   K3: Wmat = F1^T F2   (lower-triangular blocks only)
//   K4: fit  = X Wmat^T  (K-loop capped at diagonal, read-side tri-zeroing)
//   K5: avg = <x,fit> per (b,e); out = relu(x + x*(fit-avg))
// ---------------------------------------------------------------------------

typedef __attribute__((ext_vector_type(16))) __bf16       v16bf;
typedef __attribute__((ext_vector_type(8)))  float        v8f;
typedef __attribute__((ext_vector_type(8)))  unsigned int v8u;

static __device__ __forceinline__ unsigned short f32_to_bf16(float f) {
    unsigned int u = __float_as_uint(f);
    u += 0x7FFFu + ((u >> 16) & 1u);   // round-to-nearest-even
    return (unsigned short)(u >> 16);
}

// ---------------------------------------------------------------------------
// fp32 -> bf16 cast kernel, 4 elements/thread (b128 load, b64 store)
// ---------------------------------------------------------------------------
__global__ __launch_bounds__(256) void cast_bf16_kernel(
    const float* __restrict__ in, unsigned short* __restrict__ out, int n) {
    int i = (blockIdx.x * 256 + threadIdx.x) * 4;
    if (i < n) {
        float4 f = *(const float4*)(in + i);
        uint2 p;
        p.x = (unsigned int)f32_to_bf16(f.x) | ((unsigned int)f32_to_bf16(f.y) << 16);
        p.y = (unsigned int)f32_to_bf16(f.z) | ((unsigned int)f32_to_bf16(f.w) << 16);
        *(uint2*)(out + i) = p;
    }
}

// ---------------------------------------------------------------------------
// Generic bf16 GEMM: C[b][i][j] = sum_k A'(i,k) * B'(k,j)
//   A'(i,k) at A + b*aBS + i*aIs + k*aKs          (bf16)
//   B'(k,j) at B + b*bBS + k*bKs + j*bJs          (bf16)
//   C       at C + b*cBS + i*ldc + j              (f32 if CF32 else bf16)
// Template:
//   AMODE  : 0 = aKs==1 (k-contig),  1 = aIs==1 (m-contig)
//   BMODE  : 0 = bKs==1 (k-contig),  1 = bJs==1 (j-contig)
//   TRISKIP: skip whole blocks with nBase >= mBase+BM (strictly-upper dead)
//   BTRI   : lower-triangular B — K loop capped at nBase+BN and
//            B'(k,j) := 0 when k > j (never reads unwritten memory)
// Tile: 128(M) x 64(N) x 32(K), 8 waves (4x2), each wave 32x32 via 2x2 WMMA.
// All dims/bases are multiples of 8 -> b128 staging paths are 16B aligned.
// ---------------------------------------------------------------------------
#define BM 128
#define BN 64
#define BK 32

template <int AMODE, int BMODE, bool CF32, bool TRISKIP, bool BTRI>
__global__ __launch_bounds__(256) void gemm_bf16_wmma(
    const unsigned short* __restrict__ A, long long aBS, int aIs, int aKs,
    const unsigned short* __restrict__ Bm, long long bBS, int bKs, int bJs,
    void* __restrict__ Cv, long long cBS, int ldc, int K)
{
    const int nBase = blockIdx.x * BN;
    const int mBase = blockIdx.y * BM;
    const int batch = blockIdx.z;

    if (TRISKIP && nBase >= mBase + BM) return;   // strictly-upper block: dead
    const int kEnd = BTRI ? (nBase + BN) : K;

    A  += (long long)batch * aBS;
    Bm += (long long)batch * bBS;

    __shared__ alignas(16) unsigned short As[BM][BK + 2];
    __shared__ alignas(16) unsigned short Bs[BN][BK + 2];

    const int tid   = threadIdx.x;
    const int lane  = tid & 31;
    const int wave  = tid >> 5;
    const int waveM = wave >> 1;     // 0..3
    const int waveN = wave & 1;      // 0..1
    const int lm    = lane & 15;
    const int lhalf = lane >> 4;

    v8f acc[2][2];
    #pragma unroll
    for (int mi = 0; mi < 2; ++mi)
        #pragma unroll
        for (int nj = 0; nj < 2; ++nj)
            acc[mi][nj] = v8f{};

    for (int kb = 0; kb < kEnd; kb += BK) {
        __syncthreads();

        // ================= stage A tile: As[m][k] = A'(mBase+m, kb+k) =====
        if (AMODE == 0) {
            // k-contiguous: b128 loads, b32 LDS stores
            #pragma unroll
            for (int t = 0; t < 2; ++t) {
                int vi = tid + t * 256;            // 0..511 vectors of 8
                int m  = vi >> 2;
                int k0 = (vi & 3) << 3;
                uint4 d = *(const uint4*)(A + (long long)(mBase + m) * aIs + (kb + k0));
                unsigned int* dst = (unsigned int*)&As[m][k0];
                dst[0] = d.x; dst[1] = d.y; dst[2] = d.z; dst[3] = d.w;
            }
        } else {
            // m-contiguous: b128 loads along m, scatter b16 LDS stores
            #pragma unroll
            for (int t = 0; t < 2; ++t) {
                int vi = tid + t * 256;
                int k  = vi >> 4;                  // 0..31
                int m0 = (vi & 15) << 3;
                uint4 d = *(const uint4*)(A + (long long)(mBase + m0)
                                            + (long long)(kb + k) * aKs);
                unsigned int w[4] = {d.x, d.y, d.z, d.w};
                #pragma unroll
                for (int j = 0; j < 8; ++j)
                    As[m0 + j][k] = (unsigned short)(w[j >> 1] >> ((j & 1) * 16));
            }
        }

        // ========== stage B tile (N-major): Bs[n][k] = B'(kb+k, nBase+n) ==
        if (BMODE == 0) {
            if (BTRI && (kb + BK > nBase)) {
                // diagonal-straddle tile: masked scalar path (<= 2 tiles/block)
                #pragma unroll
                for (int i = tid; i < BN * BK; i += 256) {
                    int n = i >> 5, k = i & 31;
                    int kg = kb + k, jg = nBase + n;
                    unsigned short v = 0;
                    if (kg <= jg)
                        v = Bm[(long long)kg * bKs + (long long)jg * bJs];
                    Bs[n][k] = v;
                }
            } else {
                int n  = tid >> 2;
                int k0 = (tid & 3) << 3;
                uint4 d = *(const uint4*)(Bm + (kb + k0) + (long long)(nBase + n) * bJs);
                unsigned int* dst = (unsigned int*)&Bs[n][k0];
                dst[0] = d.x; dst[1] = d.y; dst[2] = d.z; dst[3] = d.w;
            }
        } else {
            int k  = tid >> 3;
            int n0 = (tid & 7) << 3;
            uint4 d = *(const uint4*)(Bm + (long long)(kb + k) * bKs + (nBase + n0));
            unsigned int w[4] = {d.x, d.y, d.z, d.w};
            #pragma unroll
            for (int j = 0; j < 8; ++j)
                Bs[n0 + j][k] = (unsigned short)(w[j >> 1] >> ((j & 1) * 16));
        }

        // prefetch next K tile into cache hierarchy (global_prefetch_b8)
        if (kb + BK < kEnd) {
            __builtin_prefetch(&A [(long long)mBase * aIs + (long long)(kb + BK) * aKs + tid], 0, 3);
            __builtin_prefetch(&Bm[(long long)(kb + BK) * bKs + (long long)nBase * bJs + tid], 0, 3);
        }
        __syncthreads();

        // ---- gather WMMA fragments per documented CDNA5 16-bit layouts ----
        // A 16x32: lane m=lane%16, half=lane/16; VGPR v pair at
        //          k = (v>=4?16:0) + half*8 + (v%4)*2
        v16bf aF[2];
        #pragma unroll
        for (int mi = 0; mi < 2; ++mi) {
            const unsigned int* row =
                (const unsigned int*)&As[waveM * 32 + mi * 16 + lm][0];
            v8u t;
            #pragma unroll
            for (int v = 0; v < 8; ++v) {
                int k = ((v >> 2) << 4) + (lhalf << 3) + ((v & 3) << 1);
                t[v] = row[k >> 1];
            }
            aF[mi] = __builtin_bit_cast(v16bf, t);
        }
        // B 32x16: lane n=lane%16, half=lane/16; VGPR v pair at
        //          k = (v>=4?16:0) + half*8 + (v%4)*2
        v16bf bF[2];
        #pragma unroll
        for (int nj = 0; nj < 2; ++nj) {
            const unsigned int* row =
                (const unsigned int*)&Bs[waveN * 32 + nj * 16 + lm][0];
            v8u t;
            #pragma unroll
            for (int v = 0; v < 8; ++v) {
                int k = ((v >> 2) << 4) + (lhalf << 3) + ((v & 3) << 1);
                t[v] = row[k >> 1];
            }
            bF[nj] = __builtin_bit_cast(v16bf, t);
        }

        // ---- 2x2 WMMA ----
        #pragma unroll
        for (int mi = 0; mi < 2; ++mi)
            #pragma unroll
            for (int nj = 0; nj < 2; ++nj)
                acc[mi][nj] = __builtin_amdgcn_wmma_f32_16x16x32_bf16(
                    false, aF[mi], false, bF[nj],
                    (short)0, acc[mi][nj], false, false);
    }

    // ---- store: C VGPR r -> (M = r + 8*half, N = lane%16) ----
    #pragma unroll
    for (int mi = 0; mi < 2; ++mi) {
        #pragma unroll
        for (int nj = 0; nj < 2; ++nj) {
            #pragma unroll
            for (int r = 0; r < 8; ++r) {
                int ig = mBase + waveM * 32 + mi * 16 + (lhalf << 3) + r;
                int jg = nBase + waveN * 32 + nj * 16 + lm;
                float val = acc[mi][nj][r];
                long long off = (long long)batch * cBS + (long long)ig * ldc + jg;
                if (CF32) ((float*)Cv)[off] = val;
                else      ((unsigned short*)Cv)[off] = f32_to_bf16(val);
            }
        }
    }
}

// ---------------------------------------------------------------------------
// Epilogue: per (b,e) row: avg = sum_s x*fit; out = relu(x + x*(fit-avg))
// float4 (b128) memory path throughout.
// ---------------------------------------------------------------------------
__global__ __launch_bounds__(256) void finalize_kernel(
    const float* __restrict__ x, const float* __restrict__ fit,
    float* __restrict__ out, int S)
{
    long long base = (long long)blockIdx.x * S;
    const float4* xr4 = (const float4*)(x   + base);
    const float4* fr4 = (const float4*)(fit + base);
    float4*       or4 = (float4*)(out + base);
    const int S4 = S >> 2;

    float partial = 0.f;
    for (int s = threadIdx.x; s < S4; s += 256) {
        float4 xv = xr4[s], fv = fr4[s];
        partial += xv.x * fv.x + xv.y * fv.y + xv.z * fv.z + xv.w * fv.w;
    }

    __shared__ float red[256];
    red[threadIdx.x] = partial;
    __syncthreads();
    #pragma unroll
    for (int st = 128; st > 0; st >>= 1) {
        if (threadIdx.x < st) red[threadIdx.x] += red[threadIdx.x + st];
        __syncthreads();
    }
    float avg = red[0];

    for (int s = threadIdx.x; s < S4; s += 256) {
        float4 xv = xr4[s], fv = fr4[s], o;
        o.x = xv.x + xv.x * (fv.x - avg);
        o.y = xv.y + xv.y * (fv.y - avg);
        o.z = xv.z + xv.z * (fv.z - avg);
        o.w = xv.w + xv.w * (fv.w - avg);
        o.x = o.x > 0.f ? o.x : 0.f;
        o.y = o.y > 0.f ? o.y : 0.f;
        o.z = o.z > 0.f ? o.z : 0.f;
        o.w = o.w > 0.f ? o.w : 0.f;
        or4[s] = o;
    }
}

// ---------------------------------------------------------------------------
extern "C" void kernel_launch(void* const* d_in, const int* in_sizes, int n_in,
                              void* d_out, int out_size, void* d_ws, size_t ws_size,
                              hipStream_t stream)
{
    (void)in_sizes; (void)n_in; (void)out_size; (void)ws_size;
    const int B = 8, E = 512, S = 2048;

    const float* x  = (const float*)d_in[0];
    const float* w1 = (const float*)d_in[1];
    const float* w2 = (const float*)d_in[2];
    float* out = (float*)d_out;

    char* ws = (char*)d_ws;
    const long long xbOff   = 0;                                  // 16 MB bf16 x
    const long long w1bOff  = xbOff  + (long long)B * E * S * 2;  // 512 KB
    const long long w2bOff  = w1bOff + (long long)E * E * 2;
    const long long f1Off   = w2bOff + (long long)E * E * 2;      // 16 MB
    const long long f2Off   = f1Off  + (long long)B * E * S * 2;  // 16 MB
    const long long wmatOff = f2Off  + (long long)B * E * S * 2;  // 64 MB bf16
    const long long fitOff  = f1Off;   // f32 fit (32 MB) reuses dead F1+F2

    unsigned short* xb   = (unsigned short*)(ws + xbOff);
    unsigned short* w1b  = (unsigned short*)(ws + w1bOff);
    unsigned short* w2b  = (unsigned short*)(ws + w2bOff);
    unsigned short* F1   = (unsigned short*)(ws + f1Off);
    unsigned short* F2   = (unsigned short*)(ws + f2Off);
    unsigned short* Wmat = (unsigned short*)(ws + wmatOff);
    float*          fitF = (float*)(ws + fitOff);

    const long long xN = (long long)B * E * S;              // 8388608
    cast_bf16_kernel<<<(int)(xN / 1024), 256, 0, stream>>>(x,  xb,  (int)xN);
    cast_bf16_kernel<<<(E * E) / 1024,   256, 0, stream>>>(w1, w1b, E * E);
    cast_bf16_kernel<<<(E * E) / 1024,   256, 0, stream>>>(w2, w2b, E * E);

    const long long esS = (long long)E * S;   // batch stride of [E][S]
    const long long sS  = (long long)S * S;   // batch stride of [S][S]
    dim3 gProj(S / BN, E / BM, B);            // (32, 4, 8)
    dim3 gWmat(S / BN, S / BM, B);            // (32, 16, 8)

    // K1: F1[f][s] = sum_e W1[e][f]*x[e][s]  (A m-contig, B j-contig)
    gemm_bf16_wmma<1, 1, false, false, false><<<gProj, 256, 0, stream>>>(
        w1b, 0, 1, E,   xb, esS, S, 1,   F1, esS, S, E);
    // K2: F2[e][s] = sum_f W2[e][f]*x[f][s]  (A k-contig, B j-contig)
    gemm_bf16_wmma<0, 1, false, false, false><<<gProj, 256, 0, stream>>>(
        w2b, 0, E, 1,   xb, esS, S, 1,   F2, esS, S, E);
    // K3: Wmat[s][t] = sum_e F1[e][s]*F2[e][t]; skip strictly-upper blocks
    gemm_bf16_wmma<1, 1, false, true, false><<<gWmat, 256, 0, stream>>>(
        F1, esS, 1, S,  F2, esS, S, 1,   Wmat, sS, S, E);
    // K4: fit[e][s] = sum_{t<=s} x[e][t]*Wmat[s][t]; lower-tri B, capped K
    gemm_bf16_wmma<0, 0, true, false, true><<<gProj, 256, 0, stream>>>(
        xb, esS, S, 1,  Wmat, sS, 1, S,  fitF, esS, S, S);

    // K5: epilogue
    finalize_kernel<<<B * E, 256, 0, stream>>>(x, fitF, out, S);
}
